// WaveSF_trans_CLS_81312320848601
// MI455X (gfx1250) — matrix operand order repaired and checked by reading, compiled, not run on hardware
//
#include <hip/hip_runtime.h>
#include <hip/hip_bf16.h>
#include <math.h>

// ---------------------------------------------------------------------------
// MI455X (gfx1250) implementation of the WaveSF transformer block.
// Compute-bound (~165 GFLOPs vs ~40MB traffic) -> route all matmuls through
// v_wmma_f32_16x16x32_f16 (wave32, 16x16x32, f32 accumulate).
// Requires ~180MB of workspace in d_ws.
// ---------------------------------------------------------------------------

typedef _Float16 f16;
typedef __attribute__((ext_vector_type(16))) _Float16 v16h;
typedef __attribute__((ext_vector_type(8)))  _Float16 v8h;
typedef __attribute__((ext_vector_type(8)))  float    v8f;

constexpr int B_   = 8;
constexpr int T_   = 16;
constexpr int H_   = 14;
constexpr int W_   = 14;
constexpr int C_   = 384;
constexpr int NH_  = 6;
constexpr int DH_  = 64;
constexpr int HW_  = H_ * W_;            // 196
constexpr int THW_ = (T_ / 2) * HW_;     // 1568
constexpr int M_   = B_ * THW_;          // 12544 tokens total
constexpr int QT_  = THW_ / 16;          // 98 q-tiles per (b,h)
constexpr int KT_  = THW_ / 32;          // 49 key steps (32 keys each)
constexpr float SC_ = 0.125f;            // 1/sqrt(64)

// ---------------- WMMA helpers (CDNA5 documented operand layouts) ----------

static __device__ __forceinline__ v8f vzero() {
  v8f v = {0.f, 0.f, 0.f, 0.f, 0.f, 0.f, 0.f, 0.f};
  return v;
}

static __device__ __forceinline__ v16h ld16g(const f16* p0, const f16* p1) {
  v8h a = *(const v8h*)p0;
  v8h b = *(const v8h*)p1;
  v16h r;
#pragma unroll
  for (int i = 0; i < 8; ++i) { r[i] = a[i]; r[i + 8] = b[i]; }
  return r;
}

// A-operand (16x32 f16): lane row m, half-wave hi: K = {hi*8..hi*8+7} u {16+hi*8..}
static __device__ __forceinline__ v16h ldA(const f16* row, int hi) {
  return ld16g(row + hi * 8, row + 16 + hi * 8);
}
// B-operand (32x16 f16): lane col n, half-wave hi: K = {hi*8..hi*8+7} u {16+hi*8..}
static __device__ __forceinline__ v16h ldB(const f16* row, int hi) {
  return ld16g(row + hi * 8, row + 16 + hi * 8);
}

static __device__ __forceinline__ v8f wmma16(v16h a, v16h b, v8f c) {
  return __builtin_amdgcn_wmma_f32_16x16x32_f16(false, a, false, b, (short)0, c,
                                                false, false);
}

// ---------------- GEMM: out = A(f16,MxK) @ W(f16,NxK)^T + bias -------------
// EPI: 0 = f16 store, 1 = GELU(exact)->f16, 2 = f32 store, 3 = res + sgn*acc
template <int EPI>
__global__ __launch_bounds__(128, 1)
void k_gemm(const f16* __restrict__ A, const f16* __restrict__ Wt,
            const float* __restrict__ bias, int N, int K,
            float* __restrict__ outF, f16* __restrict__ outH,
            const float* __restrict__ res, float sgn) {
  const int lane = threadIdx.x & 31;
  const int wv   = threadIdx.x >> 5;
  const int l15  = lane & 15;
  const int hi   = lane >> 4;
  const int n0   = blockIdx.x * 64;
  const int m0   = blockIdx.y * 64 + wv * 16;

  const f16* ap = A + (size_t)(m0 + l15) * K;
  const f16* wp[4];
#pragma unroll
  for (int j = 0; j < 4; ++j)
    wp[j] = Wt + (size_t)(n0 + j * 16 + l15) * K;

  v8f acc[4];
#pragma unroll
  for (int j = 0; j < 4; ++j) acc[j] = vzero();

  for (int k = 0; k < K; k += 32) {
    v16h a = ldA(ap + k, hi);
    __builtin_prefetch(ap + k + 256, 0, 0);   // global_prefetch_b8
#pragma unroll
    for (int j = 0; j < 4; ++j) {
      v16h b = ldB(wp[j] + k, hi);
      acc[j] = wmma16(a, b, acc[j]);
    }
  }

  const int rb = hi * 8;  // C/D layout: lanes 0-15 -> rows 0..7, 16-31 -> 8..15
#pragma unroll
  for (int j = 0; j < 4; ++j) {
    const int n  = n0 + j * 16 + l15;
    const float bn = bias[n];
#pragma unroll
    for (int r = 0; r < 8; ++r) {
      const size_t o = (size_t)(m0 + rb + r) * N + n;
      const float v = acc[j][r] + bn;
      if (EPI == 0) {
        outH[o] = (f16)v;
      } else if (EPI == 1) {
        outH[o] = (f16)(0.5f * v * (1.0f + erff(v * 0.70710678118654752f)));
      } else if (EPI == 2) {
        outF[o] = v;
      } else {
        outF[o] = res[o] + sgn * v;
      }
    }
  }
}

// ---------------- Flash attention: one wave per 16-query tile --------------
// Q,K: [B*NH][THW][64] f16 row-major.  Vt: [B*NH][64][THW] f16 (transposed).
// O: [B][THW][C] f32 (head h occupies cols h*64..h*64+63).
__global__ __launch_bounds__(128, 1)
void k_attn(const f16* __restrict__ Q, const f16* __restrict__ Kt,
            const f16* __restrict__ Vt, float* __restrict__ O) {
  __shared__ __align__(16) f16 Pl[4][16][40];  // padded: 80B rows, 16B aligned

  const int lane = threadIdx.x & 31;
  const int wv   = threadIdx.x >> 5;
  const int l15  = lane & 15;
  const int hi   = lane >> 4;
  const int rb   = hi * 8;

  const int tile = blockIdx.x * 4 + wv;
  const int bh = tile / QT_;
  const int qt = tile % QT_;
  const int b = bh / NH_, h = bh % NH_;
  const int q0 = qt * 16;

  const f16* Qb = Q  + (size_t)bh * THW_ * DH_;
  const f16* Kb = Kt + (size_t)bh * THW_ * DH_;
  const f16* Vb = Vt + (size_t)bh * DH_ * THW_;

  const f16* qp = Qb + (size_t)(q0 + l15) * DH_;
  const v16h aq0 = ldA(qp, hi);
  const v16h aq1 = ldA(qp + 32, hi);

  v8f o0 = vzero(), o1 = vzero(), o2 = vzero(), o3 = vzero();
  float mr[8], lr[8];
#pragma unroll
  for (int r = 0; r < 8; ++r) { mr[r] = -1e30f; lr[r] = 0.f; }

  f16* pw = &Pl[wv][0][0];

  for (int kt = 0; kt < KT_; ++kt) {
    const int kk = kt * 32;
    // ---- S = Q K^T for 32 keys (two 16-key subtiles, DH=64 -> 2 k-steps)
    const f16* kp0 = Kb + (size_t)(kk + l15) * DH_;
    const f16* kp1 = kp0 + 16 * DH_;
    v8f s0 = vzero(), s1 = vzero();
    s0 = wmma16(aq0, ldB(kp0, hi), s0);
    s0 = wmma16(aq1, ldB(kp0 + 32, hi), s0);
    s1 = wmma16(aq0, ldB(kp1, hi), s1);
    s1 = wmma16(aq1, ldB(kp1 + 32, hi), s1);

    // ---- online softmax (row stats live across the 16-lane C-layout group)
    float nm[8];
#pragma unroll
    for (int r = 0; r < 8; ++r) {
      s0[r] *= SC_; s1[r] *= SC_;
      nm[r] = fmaxf(s0[r], s1[r]);
    }
#pragma unroll
    for (int d = 1; d < 16; d <<= 1)
#pragma unroll
      for (int r = 0; r < 8; ++r) nm[r] = fmaxf(nm[r], __shfl_xor(nm[r], d, 32));

    float ps[8], al[8];
#pragma unroll
    for (int r = 0; r < 8; ++r) {
      const float mn = fmaxf(mr[r], nm[r]);
      al[r] = __expf(mr[r] - mn);
      mr[r] = mn;
      s0[r] = __expf(s0[r] - mn);
      s1[r] = __expf(s1[r] - mn);
      ps[r] = s0[r] + s1[r];
    }
#pragma unroll
    for (int d = 1; d < 16; d <<= 1)
#pragma unroll
      for (int r = 0; r < 8; ++r) ps[r] += __shfl_xor(ps[r], d, 32);
#pragma unroll
    for (int r = 0; r < 8; ++r) {
      lr[r] = lr[r] * al[r] + ps[r];
      o0[r] *= al[r]; o1[r] *= al[r]; o2[r] *= al[r]; o3[r] *= al[r];
    }

    // ---- transpose P (C-layout -> A-layout) through LDS (in-order per wave)
#pragma unroll
    for (int r = 0; r < 8; ++r) {
      pw[(rb + r) * 40 + l15]      = (f16)s0[r];
      pw[(rb + r) * 40 + 16 + l15] = (f16)s1[r];
    }
    const f16* pr = pw + l15 * 40;
    const v16h apk = ldA(pr, hi);

    // ---- O += P @ V  (Vt rows are contiguous in key -> clean B loads)
    const f16* vp = Vb + (size_t)l15 * THW_ + kk;
    o0 = wmma16(apk, ldB(vp, hi), o0); vp += (size_t)16 * THW_;
    o1 = wmma16(apk, ldB(vp, hi), o1); vp += (size_t)16 * THW_;
    o2 = wmma16(apk, ldB(vp, hi), o2); vp += (size_t)16 * THW_;
    o3 = wmma16(apk, ldB(vp, hi), o3);
  }

  float* Ob = O + ((size_t)b * THW_ + q0 + rb) * C_ + h * DH_ + l15;
#pragma unroll
  for (int r = 0; r < 8; ++r) {
    const float inv = 1.0f / lr[r];
    float* orow = Ob + (size_t)r * C_;
    orow[0]  = o0[r] * inv;
    orow[16] = o1[r] * inv;
    orow[32] = o2[r] * inv;
    orow[48] = o3[r] * inv;
  }
}

// ---------------- split (even/odd frames) + LayerNorm ----------------------
__global__ __launch_bounds__(128)
void k_prep_ln(const float* __restrict__ x_in, const float* __restrict__ g,
               const float* __restrict__ bt, float* __restrict__ outF,
               f16* __restrict__ outH, int parity) {
  __shared__ float sh1[128], sh2[128];
  const int tok = blockIdx.x;
  const int b = tok / THW_, t2 = tok % THW_;
  const int f = t2 / HW_, hw = t2 % HW_;
  const int t = 2 * f + parity;
  const float* src =
      x_in + ((size_t)b * (1 + T_ * HW_) + 1 + (size_t)t * HW_ + hw) * C_;

  const float x0 = src[threadIdx.x];
  const float x1 = src[threadIdx.x + 128];
  const float x2 = src[threadIdx.x + 256];
  sh1[threadIdx.x] = x0 + x1 + x2;
  sh2[threadIdx.x] = x0 * x0 + x1 * x1 + x2 * x2;
  __syncthreads();
  for (int s = 64; s > 0; s >>= 1) {
    if (threadIdx.x < s) {
      sh1[threadIdx.x] += sh1[threadIdx.x + s];
      sh2[threadIdx.x] += sh2[threadIdx.x + s];
    }
    __syncthreads();
  }
  const float mean = sh1[0] * (1.0f / C_);
  const float var  = sh2[0] * (1.0f / C_) - mean * mean;
  const float rstd = rsqrtf(var + 1e-5f);
  float* dstF = outF + (size_t)tok * C_;
  f16*   dstH = outH + (size_t)tok * C_;
#pragma unroll
  for (int i = 0; i < 3; ++i) {
    const int c = threadIdx.x + i * 128;
    const float y = (src[c] - mean) * rstd * g[c] + bt[c];
    dstF[c] = y;
    dstH[c] = (f16)y;
  }
}

// ---------------- LayerNorm f32 -> f16 -------------------------------------
__global__ __launch_bounds__(128)
void k_ln(const float* __restrict__ in, const float* __restrict__ g,
          const float* __restrict__ bt, f16* __restrict__ outH) {
  __shared__ float sh1[128], sh2[128];
  const int tok = blockIdx.x;
  const float* src = in + (size_t)tok * C_;
  const float x0 = src[threadIdx.x];
  const float x1 = src[threadIdx.x + 128];
  const float x2 = src[threadIdx.x + 256];
  sh1[threadIdx.x] = x0 + x1 + x2;
  sh2[threadIdx.x] = x0 * x0 + x1 * x1 + x2 * x2;
  __syncthreads();
  for (int s = 64; s > 0; s >>= 1) {
    if (threadIdx.x < s) {
      sh1[threadIdx.x] += sh1[threadIdx.x + s];
      sh2[threadIdx.x] += sh2[threadIdx.x + s];
    }
    __syncthreads();
  }
  const float mean = sh1[0] * (1.0f / C_);
  const float var  = sh2[0] * (1.0f / C_) - mean * mean;
  const float rstd = rsqrtf(var + 1e-5f);
  f16* dstH = outH + (size_t)tok * C_;
#pragma unroll
  for (int i = 0; i < 3; ++i) {
    const int c = threadIdx.x + i * 128;
    dstH[c] = (f16)((src[c] - mean) * rstd * g[c] + bt[c]);
  }
}

// ---------------- elementwise helpers --------------------------------------
__global__ void k_cvt(const float* __restrict__ s, f16* __restrict__ d, int n) {
  for (int i = blockIdx.x * blockDim.x + threadIdx.x; i < n;
       i += gridDim.x * blockDim.x)
    d[i] = (f16)s[i];
}
__global__ void k_addcvt(const float* __restrict__ a, const float* __restrict__ b,
                         f16* __restrict__ d, int n) {
  for (int i = blockIdx.x * blockDim.x + threadIdx.x; i < n;
       i += gridDim.x * blockDim.x)
    d[i] = (f16)(a[i] + b[i]);
}

// ---------------- head scatters --------------------------------------------
__global__ void k_scat_q(const f16* __restrict__ g, f16* __restrict__ q, int n) {
  for (int i = blockIdx.x * blockDim.x + threadIdx.x; i < n;
       i += gridDim.x * blockDim.x) {
    const int m = i / C_, c = i % C_;
    const int b = m / THW_, t = m % THW_;
    const int h = c / DH_, d = c % DH_;
    q[((size_t)(b * NH_ + h) * THW_ + t) * DH_ + d] = g[i];
  }
}
__global__ void k_scat_kv(const f16* __restrict__ g, f16* __restrict__ kH,
                          f16* __restrict__ vT, float* __restrict__ vF, int n) {
  for (int i = blockIdx.x * blockDim.x + threadIdx.x; i < n;
       i += gridDim.x * blockDim.x) {
    const int m = i / (2 * C_), c = i % (2 * C_);
    const int s = c / C_, cc = c % C_;
    const int b = m / THW_, t = m % THW_;
    const int h = cc / DH_, d = cc % DH_;
    const size_t bh = (size_t)b * NH_ + h;
    const f16 v = g[i];
    if (s == 0) kH[(bh * THW_ + t) * DH_ + d] = v;
    else {
      vT[(bh * DH_ + d) * THW_ + t] = v;
      vF[(size_t)m * C_ + cc] = (float)v;
    }
  }
}
__global__ void k_scat_qkv(const f16* __restrict__ g, f16* __restrict__ qH,
                           f16* __restrict__ kH, f16* __restrict__ vT, int n) {
  for (int i = blockIdx.x * blockDim.x + threadIdx.x; i < n;
       i += gridDim.x * blockDim.x) {
    const int m = i / (3 * C_), c = i % (3 * C_);
    const int s = c / C_, cc = c % C_;
    const int b = m / THW_, t = m % THW_;
    const int h = cc / DH_, d = cc % DH_;
    const size_t bh = (size_t)b * NH_ + h;
    const f16 v = g[i];
    if (s == 0)      qH[(bh * THW_ + t) * DH_ + d] = v;
    else if (s == 1) kH[(bh * THW_ + t) * DH_ + d] = v;
    else             vT[(bh * DH_ + d) * THW_ + t] = v;
  }
}

// ---------------- outputs ---------------------------------------------------
__global__ void k_outd(const float* __restrict__ xd, const float* __restrict__ wc,
                       float* __restrict__ out) {
  const int i = blockIdx.x * blockDim.x + threadIdx.x;
  if (i >= B_ * C_) return;
  const int b = i / C_, c = i % C_;
  const float* p = xd + (size_t)b * THW_ * C_ + c;
  float s = 0.f;
  for (int t = 0; t < THW_; ++t) s += p[(size_t)t * C_] * wc[t];
  out[i] = s;
}

__global__ __launch_bounds__(256)
void k_pool(const float* __restrict__ x_in, const float* __restrict__ xa,
            float* __restrict__ out) {
  __shared__ float sh[256];
  const int b = blockIdx.x / HW_, hw = blockIdx.x % HW_;
  float acc = 0.f;
  for (int i = threadIdx.x; i < (T_ / 2) * C_; i += 256) {
    const int t2 = i / C_, c = i % C_;
    const size_t r0 =
        ((size_t)b * (1 + T_ * HW_) + 1 + (size_t)(2 * t2) * HW_ + hw) * C_ + c;
    const size_t r1 = r0 + (size_t)HW_ * C_;
    const float xs = 0.5f * (x_in[r0] + x_in[r1]);
    const float va = xa[((size_t)b * THW_ + (size_t)t2 * HW_ + hw) * C_ + c];
    acc += xs - va;
  }
  sh[threadIdx.x] = acc;
  __syncthreads();
  for (int s = 128; s > 0; s >>= 1) {
    if (threadIdx.x < s) sh[threadIdx.x] += sh[threadIdx.x + s];
    __syncthreads();
  }
  if (threadIdx.x == 0)
    out[blockIdx.x] = sh[0] * (1.0f / ((T_ / 2) * C_));
}

// ---------------------------------------------------------------------------
extern "C" void kernel_launch(void* const* d_in, const int* in_sizes, int n_in,
                              void* d_out, int out_size, void* d_ws,
                              size_t ws_size, hipStream_t stream) {
  (void)in_sizes; (void)n_in; (void)out_size; (void)ws_size;

  const float* x_in  = (const float*)d_in[0];
  const float* Wqd   = (const float*)d_in[1];  const float* bqd  = (const float*)d_in[2];
  const float* Wkva  = (const float*)d_in[3];  const float* bkva = (const float*)d_in[4];
  const float* Wqa   = (const float*)d_in[5];  const float* bqa  = (const float*)d_in[6];
  const float* Wkvd  = (const float*)d_in[7];  const float* bkvd = (const float*)d_in[8];
  const float* Wpd   = (const float*)d_in[9];  const float* bpd  = (const float*)d_in[10];
  const float* Wpa   = (const float*)d_in[11]; const float* bpa  = (const float*)d_in[12];
  const float* gxa   = (const float*)d_in[13]; const float* bxa  = (const float*)d_in[14];
  const float* gxd   = (const float*)d_in[15]; const float* bxd  = (const float*)d_in[16];
  const float* gn1   = (const float*)d_in[17]; const float* bn1  = (const float*)d_in[18];
  const float* gn2   = (const float*)d_in[19]; const float* bn2  = (const float*)d_in[20];
  const float* Wqkv  = (const float*)d_in[21]; const float* bqkv = (const float*)d_in[22];
  const float* Wproj = (const float*)d_in[23]; const float* bproj= (const float*)d_in[24];
  const float* Wm1   = (const float*)d_in[25]; const float* bm1  = (const float*)d_in[26];
  const float* Wm2   = (const float*)d_in[27]; const float* bm2  = (const float*)d_in[28];
  const float* Wcomp = (const float*)d_in[29];

  // ---- workspace carve-up (~180MB) ----
  char* base = (char*)d_ws;
  size_t off = 0;
  auto carve = [&](size_t bytes) -> char* {
    char* p = base + off;
    off = (off + bytes + 255) & ~(size_t)255;
    return p;
  };
  float* xaF = (float*)carve((size_t)M_ * C_ * 4);      // LN'd x_a (then updated)
  float* xdF = (float*)carve((size_t)M_ * C_ * 4);      // LN'd x_d (then updated)
  float* vF  = (float*)carve((size_t)M_ * C_ * 4);      // V, token-major f32
  float* oF  = (float*)carve((size_t)M_ * C_ * 4);      // attention output
  f16* aH = (f16*)carve((size_t)M_ * C_ * 2);           // f16 of LN'd x_a
  f16* dH = (f16*)carve((size_t)M_ * C_ * 2);           // f16 of (LN'd/updated) x_d
  f16* tH = (f16*)carve((size_t)M_ * C_ * 2);           // generic f16 activation
  f16* qH = (f16*)carve((size_t)M_ * C_ * 2);           // per-head Q
  f16* kH = (f16*)carve((size_t)M_ * C_ * 2);           // per-head K
  f16* vT = (f16*)carve((size_t)M_ * C_ * 2);           // per-head V^T
  f16* U  = (f16*)carve((size_t)M_ * 4 * C_ * 2);       // GEMM-out / MLP hidden union
  f16* wH = (f16*)carve((size_t)20 * C_ * C_ * 2);      // all weights in f16

  const int wn = C_ * C_;
  f16 *wQD = wH,            *wKVA = wH + (size_t)wn,      *wQA  = wH + (size_t)3 * wn,
      *wKVD = wH + (size_t)4 * wn, *wPD = wH + (size_t)6 * wn, *wPA = wH + (size_t)7 * wn,
      *wQKV = wH + (size_t)8 * wn, *wPROJ = wH + (size_t)11 * wn,
      *wM1 = wH + (size_t)12 * wn, *wM2 = wH + (size_t)16 * wn;

  auto cg = [](int n) { return dim3((unsigned)((n + 255) / 256)); };

  // ---- weights to f16 ----
  k_cvt<<<cg(wn), 256, 0, stream>>>(Wqd, wQD, wn);
  k_cvt<<<cg(2 * wn), 256, 0, stream>>>(Wkva, wKVA, 2 * wn);
  k_cvt<<<cg(wn), 256, 0, stream>>>(Wqa, wQA, wn);
  k_cvt<<<cg(2 * wn), 256, 0, stream>>>(Wkvd, wKVD, 2 * wn);
  k_cvt<<<cg(wn), 256, 0, stream>>>(Wpd, wPD, wn);
  k_cvt<<<cg(wn), 256, 0, stream>>>(Wpa, wPA, wn);
  k_cvt<<<cg(3 * wn), 256, 0, stream>>>(Wqkv, wQKV, 3 * wn);
  k_cvt<<<cg(wn), 256, 0, stream>>>(Wproj, wPROJ, wn);
  k_cvt<<<cg(4 * wn), 256, 0, stream>>>(Wm1, wM1, 4 * wn);
  k_cvt<<<cg(4 * wn), 256, 0, stream>>>(Wm2, wM2, 4 * wn);

  // ---- split + LN ----
  k_prep_ln<<<M_, 128, 0, stream>>>(x_in, gxa, bxa, xaF, aH, 0);
  k_prep_ln<<<M_, 128, 0, stream>>>(x_in, gxd, bxd, xdF, dH, 1);

  const dim3 gB(128);
  const dim3 gridC(C_ / 64, M_ / 64);
  const dim3 grid2C(2 * C_ / 64, M_ / 64);
  const dim3 grid3C(3 * C_ / 64, M_ / 64);
  const dim3 grid4C(4 * C_ / 64, M_ / 64);
  const int nMC = M_ * C_;
  const dim3 attnGrid((B_ * NH_ * QT_) / 4);

  // ---- cross-attn 1: q = x_d, kv = x_a; x_d -= lin(o+v, Wpd) ----
  k_gemm<0><<<gridC, gB, 0, stream>>>(dH, wQD, bqd, C_, C_, nullptr, U, nullptr, 0.f);
  k_scat_q<<<cg(nMC), 256, 0, stream>>>(U, qH, nMC);
  k_gemm<0><<<grid2C, gB, 0, stream>>>(aH, wKVA, bkva, 2 * C_, C_, nullptr, U, nullptr, 0.f);
  k_scat_kv<<<cg(2 * nMC), 256, 0, stream>>>(U, kH, vT, vF, 2 * nMC);
  k_attn<<<attnGrid, gB, 0, stream>>>(qH, kH, vT, oF);
  k_addcvt<<<cg(nMC), 256, 0, stream>>>(oF, vF, tH, nMC);
  k_gemm<3><<<gridC, gB, 0, stream>>>(tH, wPD, bpd, C_, C_, xdF, nullptr, xdF, -1.0f);

  // ---- cross-attn 2: q = x_a, kv = x_d(updated); x_a += lin(o+v, Wpa) ----
  k_cvt<<<cg(nMC), 256, 0, stream>>>(xdF, dH, nMC);
  k_gemm<0><<<gridC, gB, 0, stream>>>(aH, wQA, bqa, C_, C_, nullptr, U, nullptr, 0.f);
  k_scat_q<<<cg(nMC), 256, 0, stream>>>(U, qH, nMC);
  k_gemm<0><<<grid2C, gB, 0, stream>>>(dH, wKVD, bkvd, 2 * C_, C_, nullptr, U, nullptr, 0.f);
  k_scat_kv<<<cg(2 * nMC), 256, 0, stream>>>(U, kH, vT, vF, 2 * nMC);
  k_attn<<<attnGrid, gB, 0, stream>>>(qH, kH, vT, oF);
  k_addcvt<<<cg(nMC), 256, 0, stream>>>(oF, vF, tH, nMC);
  k_gemm<3><<<gridC, gB, 0, stream>>>(tH, wPA, bpa, C_, C_, xaF, nullptr, xaF, 1.0f);

  // ---- self-attn on x_d ----
  k_ln<<<M_, 128, 0, stream>>>(xdF, gn1, bn1, tH);
  k_gemm<0><<<grid3C, gB, 0, stream>>>(tH, wQKV, bqkv, 3 * C_, C_, nullptr, U, nullptr, 0.f);
  k_scat_qkv<<<cg(3 * nMC), 256, 0, stream>>>(U, qH, kH, vT, 3 * nMC);
  k_attn<<<attnGrid, gB, 0, stream>>>(qH, kH, vT, oF);
  k_cvt<<<cg(nMC), 256, 0, stream>>>(oF, tH, nMC);
  k_gemm<3><<<gridC, gB, 0, stream>>>(tH, wPROJ, bproj, C_, C_, xdF, nullptr, xdF, 1.0f);

  // ---- MLP: x_d = W2 @ gelu(W1 @ ln(x_d)) ----
  k_ln<<<M_, 128, 0, stream>>>(xdF, gn2, bn2, tH);
  k_gemm<1><<<grid4C, gB, 0, stream>>>(tH, wM1, bm1, 4 * C_, C_, nullptr, U, nullptr, 0.f);
  k_gemm<2><<<gridC, gB, 0, stream>>>(U, wM2, bm2, C_, 4 * C_, xdF, nullptr, nullptr, 0.f);

  // ---- outputs ----
  k_outd<<<cg(B_ * C_), 256, 0, stream>>>(xdF, Wcomp, (float*)d_out);
  k_pool<<<dim3(B_ * HW_), 256, 0, stream>>>(x_in, xaF, (float*)d_out + B_ * C_);
}
